// SelfAttention_40785009443071
// MI455X (gfx1250) — matrix operand order repaired and checked by reading, not compiled, not run on hardware
//
#include <hip/hip_runtime.h>
#include <stdint.h>

// ---------------------------------------------------------------------------
// Self-attention (S=2048, B=8, D=512), fp32 in/out, bf16 WMMA everywhere.
// Workspace (64 MiB): Qb bf16[B][S][D] | Kb bf16[B][S][D] | Vt bf16[B][D][S]
//                     | Sc f32[S][S] (per batch; softmax rewrites rows in
//                       place as bf16 probs, row stride 4096 halves).
// QK^T and P*V use a double-buffered GLOBAL_LOAD_ASYNC_TO_LDS pipeline
// (ASYNCcnt) — builtin confirmed present by the round-2 diagnostic.
// ---------------------------------------------------------------------------

#if defined(__has_builtin)
#if __has_builtin(__builtin_amdgcn_global_load_async_to_lds_b128) && \
    __has_builtin(__builtin_amdgcn_s_wait_asynccnt)
#define HAVE_ASYNC_LDS 1
#endif
#endif

typedef __attribute__((ext_vector_type(16))) __bf16 v16bf;
typedef __attribute__((ext_vector_type(8)))  float  v8f;
typedef int v4i __attribute__((vector_size(16)));   // matches builtin param type

constexpr int S_LEN = 2048, BATCH = 8, DIM = 512;
constexpr int BM = 128, BN = 128, BK = 32;
constexpr int LDT = 40;                        // padded LDS row stride (halves)
constexpr float SCALE = 0.04419417382415922f;  // 1/sqrt(512)

__device__ __forceinline__ unsigned short f2bf(float f) {
  unsigned int u = __builtin_bit_cast(unsigned int, f);
  u += 0x7FFFu + ((u >> 16) & 1u);             // round-to-nearest-even
  return (unsigned short)(u >> 16);
}

struct TwoQ { uint4 a, b; };
__device__ __forceinline__ v16bf frag2(const unsigned short* p0,
                                       const unsigned short* p1) {
  TwoQ t;
  t.a = *(const uint4*)p0;
  t.b = *(const uint4*)p1;
  return __builtin_bit_cast(v16bf, t);
}

#ifdef HAVE_ASYNC_LDS
// Builtin expects v4i* operands (per hipcc diagnostic). AS-qualified pointers
// implicitly convert to generic if needed (generic is a superset), so these
// casts are safe for either declared address space.
#define GPTR(p) ((__attribute__((address_space(1))) v4i*)(uintptr_t)(p))
#define LPTR(p) ((__attribute__((address_space(3))) v4i*)(uint32_t)(uintptr_t)(p))
// Copy 32 contiguous bytes global->LDS via the async unit. The instruction
// IOFFSET is added to BOTH the global and the LDS address (ISA 08 §4.4), so
// one address setup serves both 16B transfers.
__device__ __forceinline__ void async_cp32(const void* g, void* l) {
  __builtin_amdgcn_global_load_async_to_lds_b128(GPTR(g), LPTR(l), 0, 0);
  __builtin_amdgcn_global_load_async_to_lds_b128(GPTR(g), LPTR(l), 16, 0);
}
#endif

// Per-wave inner step: one A-frag (16xK rows of this wave), 8 B tiles across
// N. B-frags are pre-loaded into registers so ds_loads can be batched ahead
// of the WMMA chain instead of wait-per-WMMA.
#define WMMA_TILE_COMPUTE(pA, pB, acc, wave, lh, hi)                          \
  do {                                                                        \
    const unsigned short* _ap = &(pA)[(16 * (wave) + (lh)) * LDT + (hi) * 8]; \
    v16bf _a = frag2(_ap, _ap + 16);                                          \
    v16bf _b[8];                                                              \
    _Pragma("unroll")                                                         \
    for (int _t = 0; _t < 8; ++_t) {                                          \
      const unsigned short* _bp = &(pB)[(16 * _t + (lh)) * LDT + (hi) * 8];   \
      _b[_t] = frag2(_bp, _bp + 16);                                          \
    }                                                                         \
    _Pragma("unroll")                                                         \
    for (int _t = 0; _t < 8; ++_t)                                            \
      acc[_t] = __builtin_amdgcn_wmma_f32_16x16x32_bf16(                      \
          false, _a, false, _b[_t], (short)0, acc[_t], false, false);         \
  } while (0)

// ---------------------------------------------------------------------------
// Stage 1: Q/K/V projections. Y = X @ W^T (f32 source -> bf16 conversion in
// the staging step, so this stays on the synchronous path).
// blockIdx.z = b*3 + mat (0=Q, 1=K, 2=V). V stored transposed [d][s].
// ---------------------------------------------------------------------------
__global__ __launch_bounds__(256)
void attn_qkv_proj(const float* __restrict__ x,
                   const float* __restrict__ Wq,
                   const float* __restrict__ Wk,
                   const float* __restrict__ Wv,
                   unsigned short* __restrict__ Qb,
                   unsigned short* __restrict__ Kb,
                   unsigned short* __restrict__ Vt) {
  __shared__ unsigned short sA[BM * LDT];
  __shared__ unsigned short sB[BN * LDT];
  const int tid = threadIdx.x;
  const int wave = tid >> 5, lane = tid & 31, lh = lane & 15, hi = (lane >> 4) & 1;
  const int m0 = blockIdx.x * BM;
  const int n0 = blockIdx.y * BN;
  const int b = blockIdx.z / 3, mat = blockIdx.z % 3;
  const float* W = (mat == 0) ? Wq : ((mat == 1) ? Wk : Wv);
  const int srow = tid >> 1, sseg = tid & 1;

  v8f acc[8];
  v8f zc = {0.f, 0.f, 0.f, 0.f, 0.f, 0.f, 0.f, 0.f};
  #pragma unroll
  for (int t = 0; t < 8; ++t) acc[t] = zc;

  for (int k0 = 0; k0 < DIM; k0 += BK) {
    // A tile: X[b] rows (global layout x[(s*B + b)*D + d]), f32 -> bf16
    const float* ag = x + ((size_t)(m0 + srow) * BATCH + b) * DIM + k0 + sseg * 16;
    unsigned short* ad = &sA[srow * LDT + sseg * 16];
    #pragma unroll
    for (int j = 0; j < 16; ++j) ad[j] = f2bf(ag[j]);
    // B^T tile: rows of W (W[n][k] contiguous in k), f32 -> bf16
    const float* bg = W + (size_t)(n0 + srow) * DIM + k0 + sseg * 16;
    unsigned short* bd = &sB[srow * LDT + sseg * 16];
    #pragma unroll
    for (int j = 0; j < 16; ++j) bd[j] = f2bf(bg[j]);
    __syncthreads();
    WMMA_TILE_COMPUTE(sA, sB, acc, wave, lh, hi);
    __syncthreads();
  }

  const size_t boffE = (size_t)b * S_LEN * DIM;
  if (mat == 2) {                       // V, stored transposed [d][s]
    unsigned short* dst = Vt + boffE;
    #pragma unroll
    for (int t = 0; t < 8; ++t) {
      const int n = n0 + 16 * t + lh;
      #pragma unroll
      for (int r = 0; r < 8; ++r) {
        const int m = m0 + 16 * wave + r + hi * 8;
        dst[(size_t)n * S_LEN + m] = f2bf(acc[t][r]);
      }
    }
  } else {                              // Q or K, row-major [s][d]
    unsigned short* dst = ((mat == 0) ? Qb : Kb) + boffE;
    #pragma unroll
    for (int t = 0; t < 8; ++t) {
      const int n = n0 + 16 * t + lh;
      #pragma unroll
      for (int r = 0; r < 8; ++r) {
        const int m = m0 + 16 * wave + r + hi * 8;
        dst[(size_t)m * DIM + n] = f2bf(acc[t][r]);
      }
    }
  }
}

// ---------------------------------------------------------------------------
// Unified bf16 GEMM: C[m][n] = cscale * sum_k A[m][k] * B[n][k]
// Used for scores (A=Q, B=K, C=Sc, cscale=1/sqrt(D)) and for P*V
// (A=P rows stride 4096, B=Vt rows, C=out[b], cscale=1).
// Double-buffered async global->LDS staging.
// ---------------------------------------------------------------------------
__global__ __launch_bounds__(256)
void attn_gemm_bf16(const unsigned short* __restrict__ A, int lda,
                    const unsigned short* __restrict__ B, int ldb,
                    int K, float* __restrict__ C, int ldc, float cscale) {
  __shared__ unsigned short sA[2][BM * LDT];
  __shared__ unsigned short sB[2][BN * LDT];
  const int tid = threadIdx.x;
  const int wave = tid >> 5, lane = tid & 31, lh = lane & 15, hi = (lane >> 4) & 1;
  const int m0 = blockIdx.x * BM;
  const int n0 = blockIdx.y * BN;
  const int srow = tid >> 1, sseg = tid & 1;

  const unsigned short* ga0 = A + (size_t)(m0 + srow) * lda + sseg * 16;
  const unsigned short* gb0 = B + (size_t)(n0 + srow) * ldb + sseg * 16;
  const int soff = srow * LDT + sseg * 16;

  v8f acc[8];
  v8f zc = {0.f, 0.f, 0.f, 0.f, 0.f, 0.f, 0.f, 0.f};
  #pragma unroll
  for (int t = 0; t < 8; ++t) acc[t] = zc;

#ifdef HAVE_ASYNC_LDS
  const int nK = K / BK;
  // Prologue: tile 0 in flight (4 async b128 per thread per tile).
  async_cp32(ga0, &sA[0][soff]);
  async_cp32(gb0, &sB[0][soff]);
  for (int i = 0; i < nK; ++i) {
    if (i + 1 < nK) {
      async_cp32(ga0 + (size_t)(i + 1) * BK, &sA[(i + 1) & 1][soff]);
      async_cp32(gb0 + (size_t)(i + 1) * BK, &sB[(i + 1) & 1][soff]);
      __builtin_amdgcn_s_wait_asynccnt(4);   // tile i drained, tile i+1 in flight
    } else {
      __builtin_amdgcn_s_wait_asynccnt(0);
    }
    __syncthreads();
    WMMA_TILE_COMPUTE(sA[i & 1], sB[i & 1], acc, wave, lh, hi);
    __syncthreads();
  }
#else
  for (int k0 = 0; k0 < K; k0 += BK) {
    const uint4* ag = (const uint4*)(ga0 + k0);
    uint4* ad = (uint4*)&sA[0][soff];
    ad[0] = ag[0]; ad[1] = ag[1];
    const uint4* bg = (const uint4*)(gb0 + k0);
    uint4* bd = (uint4*)&sB[0][soff];
    bd[0] = bg[0]; bd[1] = bg[1];
    __syncthreads();
    WMMA_TILE_COMPUTE(sA[0], sB[0], acc, wave, lh, hi);
    __syncthreads();
  }
#endif

  #pragma unroll
  for (int t = 0; t < 8; ++t) {
    const int n = n0 + 16 * t + lh;
    #pragma unroll
    for (int r = 0; r < 8; ++r) {
      const int m = m0 + 16 * wave + r + hi * 8;
      C[(size_t)m * ldc + n] = acc[t][r] * cscale;
    }
  }
}

// ---------------------------------------------------------------------------
// Row softmax with both key-side masks; rewrites the score row in place as
// bf16 probabilities (row q lives at ((ushort*)Sc) + q*4096).
// ---------------------------------------------------------------------------
__global__ __launch_bounds__(256)
void attn_softmax(float* __restrict__ Sc,
                  const unsigned char* __restrict__ mask,    // (B, S)
                  const unsigned char* __restrict__ punct,   // (S, B)
                  int b) {
  __shared__ float red[256];
  const int q = blockIdx.x;
  const int tid = threadIdx.x;
  float* Srow = Sc + (size_t)q * S_LEN;
  unsigned short* Prow = (unsigned short*)Sc + (size_t)q * (2 * S_LEN);

  float vals[8];
  bool ok[8];
  float mx = -INFINITY;
  #pragma unroll
  for (int j = 0; j < 8; ++j) {
    const int kk = tid + 256 * j;
    const bool masked = (mask[(size_t)b * S_LEN + kk] != 0) ||
                        (punct[(size_t)kk * BATCH + b] != 0);
    ok[j] = !masked;
    vals[j] = Srow[kk];                 // all reads land in registers first
    if (!masked && vals[j] > mx) mx = vals[j];
  }
  red[tid] = mx;
  __syncthreads();
  for (int off = 128; off > 0; off >>= 1) {
    if (tid < off && red[tid + off] > red[tid]) red[tid] = red[tid + off];
    __syncthreads();
  }
  mx = red[0];
  __syncthreads();

  float sum = 0.f;
  #pragma unroll
  for (int j = 0; j < 8; ++j) {
    const float e = ok[j] ? __expf(vals[j] - mx) : 0.f;
    vals[j] = e;
    sum += e;
  }
  red[tid] = sum;
  __syncthreads();
  for (int off = 128; off > 0; off >>= 1) {
    if (tid < off) red[tid] += red[tid + off];
    __syncthreads();
  }
  const float inv = 1.0f / red[0];
  __syncthreads();
  #pragma unroll
  for (int j = 0; j < 8; ++j) Prow[tid + 256 * j] = f2bf(vals[j] * inv);
}

// ---------------------------------------------------------------------------
extern "C" void kernel_launch(void* const* d_in, const int* in_sizes, int n_in,
                              void* d_out, int out_size, void* d_ws, size_t ws_size,
                              hipStream_t stream) {
  const float* x            = (const float*)d_in[0];
  const unsigned char* mask = (const unsigned char*)d_in[1];   // jnp.bool_ = 1 byte
  const unsigned char* punc = (const unsigned char*)d_in[2];
  const float* Wq           = (const float*)d_in[3];
  const float* Wk           = (const float*)d_in[4];
  const float* Wv           = (const float*)d_in[5];
  float* out                = (float*)d_out;

  const size_t QKV = (size_t)BATCH * S_LEN * DIM;   // elements
  unsigned short* Qb = (unsigned short*)d_ws;
  unsigned short* Kb = Qb + QKV;
  unsigned short* Vt = Kb + QKV;
  float* Sc          = (float*)(Vt + QKV);          // 2048x2048 f32, per batch

  dim3 blk(256);
  attn_qkv_proj<<<dim3(S_LEN / BM, DIM / BN, BATCH * 3), blk, 0, stream>>>(
      x, Wq, Wk, Wv, Qb, Kb, Vt);

  for (int b = 0; b < BATCH; ++b) {
    const size_t boffE = (size_t)b * S_LEN * DIM;
    // scores = SCALE * Q @ K^T   (M=N=2048, K=512)
    attn_gemm_bf16<<<dim3(S_LEN / BM, S_LEN / BN), blk, 0, stream>>>(
        Qb + boffE, DIM, Kb + boffE, DIM, DIM, Sc, S_LEN, SCALE);
    attn_softmax<<<dim3(S_LEN), dim3(256), 0, stream>>>(Sc, mask, punc, b);
    // out[b] = P @ V   (M=2048, N=512, K=2048); A = bf16 P rows, stride 4096
    attn_gemm_bf16<<<dim3(S_LEN / BM, DIM / BN), blk, 0, stream>>>(
        (const unsigned short*)Sc, 2 * S_LEN, Vt + boffE, S_LEN, S_LEN,
        out + boffE, DIM, 1.0f);
  }
}
